// LSTM_25391846654283
// MI455X (gfx1250) — matrix operand order repaired and checked by reading, not compiled, not run on hardware
//
#include <hip/hip_runtime.h>
#include <hip/hip_bf16.h>

typedef __attribute__((ext_vector_type(16))) _Float16 v16h;
typedef __attribute__((ext_vector_type(8)))  _Float16 v8h;
typedef __attribute__((ext_vector_type(8)))  float    v8f;

constexpr int kB = 512, kS = 512, kI = 256, kH = 512, kO = 256;
constexpr int LDS_STRIDE = 40;   // 32 halves + 8 pad (80B rows, 16B aligned)

// ---------------- fast activations (v_exp_f32 + v_rcp_f32) ----------------
__device__ __forceinline__ float fast_sigmoid(float x) {
  return __builtin_amdgcn_rcpf(1.0f + __expf(-x));
}
__device__ __forceinline__ float fast_tanh(float x) {
  float e = __expf(-2.0f * x);
  return (1.0f - e) * __builtin_amdgcn_rcpf(1.0f + e);
}

// ---------------- CDNA5 async global->LDS copy (ASYNCcnt-tracked) ---------
// 16 bytes per lane; LDS operand VGPR takes the workgroup-relative LDS byte
// offset (= low 32 bits of the generic shared pointer per the aperture rules).
__device__ __forceinline__ void async_copy16(const _Float16* g, _Float16* l) {
  unsigned loff = (unsigned)(uintptr_t)l;
  unsigned long long ga = (unsigned long long)(uintptr_t)g;
  asm volatile("global_load_async_to_lds_b128 %0, %1, off"
               :: "v"(loff), "v"(ga) : "memory");
}

// Fused panel synchronization:
//   s_wait_asynccnt 0 : this wave's async copies for the panel have landed
//   s_wait_dscnt 0    : this wave's in-flight LDS fragment reads are done
//                       (so the buffer we are about to overwrite is free)
//   barrier           : all waves agree
// Returns a dummy 0 in an SGPR; callers add it to the LDS pointer so every
// subsequent ds_load has a hard data dependency on this sync and can never be
// scheduled above it.
__device__ __forceinline__ unsigned sync_panel() {
  unsigned zero;
  asm volatile("s_wait_asynccnt 0x0\n\t"
               "s_wait_dscnt 0x0\n\t"
               "s_barrier_signal -1\n\t"
               "s_barrier_wait -1\n\t"
               "s_mov_b32 %0, 0"
               : "=s"(zero) :: "memory");
  return zero;
}

// Issue async staging of a 128x32 f16 A-panel (512 x 16B chunks, 2/thread),
// branch-free: rows tid>>2 and tid>>2 + 64.
__device__ __forceinline__ void issue_panel(_Float16* sbuf, const _Float16* srcA,
                                            int rs, int tid) {
  int row0 = tid >> 2, cc = (tid & 3) << 3;
  async_copy16(srcA + (size_t)row0 * rs + cc, sbuf + row0 * LDS_STRIDE + cc);
  int row1 = row0 + 64;
  async_copy16(srcA + (size_t)row1 * rs + cc, sbuf + row1 * LDS_STRIDE + cc);
}

union FragU { v16h v; v8h p[2]; };

// A fragment per ISA 16-bit A 16x32 layout:
// lane m (0..15)=row; group0 holds K {0..7,16..23}, group1 holds K {8..15,24..31}
__device__ __forceinline__ v16h load_a_frag(const _Float16* sA, int lane, int wave) {
  FragU u;
  const _Float16* pa = sA + ((wave << 4) + (lane & 15)) * LDS_STRIDE + ((lane >> 4) << 3);
  u.p[0] = *(const v8h*)pa;
  u.p[1] = *(const v8h*)(pa + 16);
  return u.v;
}
// B fragment: weights pre-packed so lane's 16 halves are contiguous (32B)
__device__ __forceinline__ v16h load_b_frag(const _Float16* pb) {
  FragU u;
  u.p[0] = *(const v8h*)pb;
  u.p[1] = *(const v8h*)(pb + 8);
  return u.v;
}

// ---------------- weight packing: W[N][K] f32 -> WMMA B-fragment order -----
// dst layout: [ntile=N/16][kpanel=K/32][lane=32][16 halves], where
// lane&15 = column n within tile, lane>>4 selects K {0..7,16..23}/{8..15,24..31}.
__global__ void pack_w_kernel(const float* __restrict__ src, _Float16* __restrict__ dst,
                              int N, int K) {
  int ntiles = N >> 4, kpan = K >> 5;
  int idx = blockIdx.x * blockDim.x + threadIdx.x;
  int total = ntiles * kpan * 32;
  if (idx >= total) return;
  int lane = idx & 31;
  int kp = (idx >> 5) % kpan;
  int tn = idx / (kpan * 32);
  int n = (tn << 4) + (lane & 15);
  int kbase = (kp << 5) + ((lane >> 4) << 3);
  const float* s = src + (size_t)n * K + kbase;
  _Float16* d = dst + ((size_t)idx << 4);
#pragma unroll
  for (int j = 0; j < 16; ++j) d[j] = (_Float16)s[j + ((j >> 3) << 3)];
}

__global__ void addvec_kernel(const float* __restrict__ a, const float* __restrict__ b,
                              float* __restrict__ d, int n) {
  int i = blockIdx.x * blockDim.x + threadIdx.x;
  if (i < n) d[i] = a[i] + b[i];
}

// x[b][t][i] f32 -> xT[t][b][i] f16 (one streaming pass; halves step bandwidth)
__global__ void xpose_kernel(const float* __restrict__ x, _Float16* __restrict__ xT) {
  size_t idx = (size_t)blockIdx.x * blockDim.x + threadIdx.x; // chunk of 8 along i
  int ic = (int)(idx & 31);          // 256/8 = 32 chunks per (t,b) row
  size_t bt = idx >> 5;              // t*kB + b
  int b = (int)(bt & (kB - 1));
  int t = (int)(bt >> 9);
  const float* src = x + ((size_t)b * kS + t) * kI + (ic << 3);
  _Float16* dst = xT + (bt << 8) + (ic << 3);
#pragma unroll
  for (int j = 0; j < 8; ++j) dst[j] = (_Float16)src[j];
}

__global__ void init_kernel(_Float16* __restrict__ h0, float* __restrict__ c) {
  int idx = blockIdx.x * blockDim.x + threadIdx.x;  // kB*kH threads
  h0[idx] = (_Float16)0.0f;
  c[idx] = 0.0f;
}

// ---------------- one LSTM timestep --------------------------------------
// grid = (kH/16, kB/128); 256 threads = 8 waves; wave w owns rows m0+16w..+15,
// cols n0..n0+15, for all 4 gates (4 f32 accumulators sharing one A fragment).
// A panels staged via double-buffered async global->LDS copies.
__global__ __launch_bounds__(256) void lstm_step_kernel(
    const _Float16* __restrict__ xt,    // [kB][kI]  (this timestep's slice)
    const _Float16* __restrict__ hin,   // [kB][kH]
    _Float16* __restrict__ hout,        // [kB][kH]
    float* __restrict__ cst,            // [kB][kH]
    const _Float16* __restrict__ wx,    // [4][kI/32*kH fragments]
    const _Float16* __restrict__ wh,    // [4][kH/32*kH fragments]
    const float* __restrict__ bias)     // [4][kH]  (bx+bh combined)
{
  __shared__ __align__(16) _Float16 sA[2][128 * LDS_STRIDE];
  const int tid = threadIdx.x, lane = tid & 31, wave = tid >> 5;
  const int tn = blockIdx.x, n0 = tn << 4, m0 = blockIdx.y << 7;
  v8f a0 = {}, a1 = {}, a2 = {}, a3 = {};

  // prologue: stage panel 0 (x source) into buffer 0
  issue_panel(sA[0], xt + (size_t)m0 * kI, kI, tid);

  // unified K loop: panels 0..7 from xt (K=kI), panels 8..23 from hin (K=kH)
  for (int p = 0; p < 24; ++p) {
    unsigned z = sync_panel();   // copies landed everywhere; prev buffer free
    if (p + 1 < 24) {            // overlap: stage panel p+1 while computing p
      const _Float16* nsrc; int nrs;
      if (p + 1 < 8) { nsrc = xt + (size_t)m0 * kI + ((p + 1) << 5);       nrs = kI; }
      else           { nsrc = hin + (size_t)m0 * kH + ((p + 1 - 8) << 5);  nrs = kH; }
      issue_panel(sA[(p + 1) & 1], nsrc, nrs, tid);
    }
    const _Float16* sbuf = sA[p & 1] + z;   // data-dependent on sync_panel()
    v16h af = load_a_frag(sbuf, lane, wave);
    const _Float16* pb; size_t gs;
    if (p < 8) { pb = wx + (((size_t)tn * (kI / 32) + p) * 32 + lane) * 16;
                 gs = (size_t)kI * kH; }
    else       { pb = wh + (((size_t)tn * (kH / 32) + (p - 8)) * 32 + lane) * 16;
                 gs = (size_t)kH * kH; }
    v16h b0 = load_b_frag(pb);
    v16h b1 = load_b_frag(pb + gs);
    v16h b2 = load_b_frag(pb + 2 * gs);
    v16h b3 = load_b_frag(pb + 3 * gs);
    a0 = __builtin_amdgcn_wmma_f32_16x16x32_f16(false, af, false, b0, (short)0, a0, false, false);
    a1 = __builtin_amdgcn_wmma_f32_16x16x32_f16(false, af, false, b1, (short)0, a1, false, false);
    a2 = __builtin_amdgcn_wmma_f32_16x16x32_f16(false, af, false, b2, (short)0, a2, false, false);
    a3 = __builtin_amdgcn_wmma_f32_16x16x32_f16(false, af, false, b3, (short)0, a3, false, false);
  }

  // ---- fused gate activations + state update ----
  // C/D layout: VGPR r -> row r (lanes 0-15) / row r+8 (lanes 16-31); N = lane&15
  const int col = n0 + (lane & 15);
  const int rbase = m0 + (wave << 4) + ((lane >> 4) << 3);
  const float bg = bias[col];
  const float bi = bias[kH + col];
  const float bf = bias[2 * kH + col];
  const float bo = bias[3 * kH + col];
#pragma unroll
  for (int r = 0; r < 8; ++r) {
    const size_t off = (size_t)(rbase + r) * kH + col;
    float gv = fast_tanh(a0[r] + bg);
    float iv = fast_sigmoid(a1[r] + bi);
    float fv = fast_sigmoid(a2[r] + bf);
    float ov = fast_sigmoid(a3[r] + bo);
    float cn = gv * iv + cst[off] * fv;
    cst[off] = cn;
    hout[off] = (_Float16)(fast_tanh(cn) * ov);
  }
}

// ---------------- final projection: out = h @ Wph.T + bph ------------------
__global__ __launch_bounds__(256) void proj_kernel(
    const _Float16* __restrict__ hfin,  // [kB][kH]
    const _Float16* __restrict__ wp,    // packed [kO/16][kH/32][32][16]
    const float* __restrict__ bp,       // [kO]
    float* __restrict__ out)            // [kB][kO]
{
  __shared__ __align__(16) _Float16 sA[2][128 * LDS_STRIDE];
  const int tid = threadIdx.x, lane = tid & 31, wave = tid >> 5;
  const int tn = blockIdx.x, m0 = blockIdx.y << 7;
  v8f acc = {};

  issue_panel(sA[0], hfin + (size_t)m0 * kH, kH, tid);
  for (int p = 0; p < kH / 32; ++p) {
    unsigned z = sync_panel();
    if (p + 1 < kH / 32)
      issue_panel(sA[(p + 1) & 1], hfin + (size_t)m0 * kH + ((p + 1) << 5), kH, tid);
    v16h af = load_a_frag(sA[p & 1] + z, lane, wave);
    v16h bfr = load_b_frag(wp + (((size_t)tn * (kH / 32) + p) * 32 + lane) * 16);
    acc = __builtin_amdgcn_wmma_f32_16x16x32_f16(false, af, false, bfr, (short)0, acc, false, false);
  }
  const int col = (tn << 4) + (lane & 15);
  const int rbase = m0 + (wave << 4) + ((lane >> 4) << 3);
  const float bb = bp[col];
#pragma unroll
  for (int r = 0; r < 8; ++r)
    out[(size_t)(rbase + r) * kO + col] = acc[r] + bb;
}

// ---------------- host driver ---------------------------------------------
extern "C" void kernel_launch(void* const* d_in, const int* in_sizes, int n_in,
                              void* d_out, int out_size, void* d_ws, size_t ws_size,
                              hipStream_t stream) {
  (void)in_sizes; (void)n_in; (void)out_size; (void)ws_size;
  const float* x = (const float*)d_in[0];
  const float* Wx[4] = {(const float*)d_in[1], (const float*)d_in[5],
                        (const float*)d_in[9], (const float*)d_in[13]};
  const float* bx[4] = {(const float*)d_in[2], (const float*)d_in[6],
                        (const float*)d_in[10], (const float*)d_in[14]};
  const float* Wh[4] = {(const float*)d_in[3], (const float*)d_in[7],
                        (const float*)d_in[11], (const float*)d_in[15]};
  const float* bh[4] = {(const float*)d_in[4], (const float*)d_in[8],
                        (const float*)d_in[12], (const float*)d_in[16]};
  const float* Wp = (const float*)d_in[17];
  const float* bp = (const float*)d_in[18];

  // workspace carve-up (~139.7 MB total)
  char* p = (char*)d_ws;
  _Float16* xT   = (_Float16*)p;  p += (size_t)kS * kB * kI * 2;
  _Float16* h0   = (_Float16*)p;  p += (size_t)kB * kH * 2;
  _Float16* h1   = (_Float16*)p;  p += (size_t)kB * kH * 2;
  float*    c    = (float*)p;     p += (size_t)kB * kH * 4;
  _Float16* wxp  = (_Float16*)p;  p += (size_t)4 * kI * kH * 2;
  _Float16* whp  = (_Float16*)p;  p += (size_t)4 * kH * kH * 2;
  float*    bsum = (float*)p;     p += (size_t)4 * kH * 4;
  _Float16* wpp  = (_Float16*)p;

  // pack weights into WMMA fragment order + combine biases (one-time cost)
  for (int g = 0; g < 4; ++g) {
    pack_w_kernel<<<(32 * 8 * 32 + 255) / 256, 256, 0, stream>>>(
        Wx[g], wxp + (size_t)g * kI * kH, kH, kI);
    pack_w_kernel<<<(32 * 16 * 32 + 255) / 256, 256, 0, stream>>>(
        Wh[g], whp + (size_t)g * kH * kH, kH, kH);
    addvec_kernel<<<(kH + 255) / 256, 256, 0, stream>>>(bx[g], bh[g], bsum + g * kH, kH);
  }
  pack_w_kernel<<<(16 * 16 * 32 + 255) / 256, 256, 0, stream>>>(Wp, wpp, kO, kH);

  // convert/transpose x once: f32 [B][S][I] -> f16 [S][B][I]
  xpose_kernel<<<(kS * kB * (kI / 8)) / 256, 256, 0, stream>>>(x, xT);
  init_kernel<<<(kB * kH) / 256, 256, 0, stream>>>(h0, c);

  // sequential recurrence: one launch per timestep, ping-pong h buffers
  for (int t = 0; t < kS; ++t) {
    const _Float16* hin = (t & 1) ? h1 : h0;
    _Float16* hout = (t & 1) ? h0 : h1;
    lstm_step_kernel<<<dim3(kH / 16, kB / 128), 256, 0, stream>>>(
        xT + (size_t)t * kB * kI, hin, hout, c, wxp, whp, bsum);
  }
  // after 512 steps (even), final h lives in h0
  proj_kernel<<<dim3(kO / 16, kB / 128), 256, 0, stream>>>(h0, wpp, bp, (float*)d_out);
}
